// MambaRegressor_19481971654886
// MI455X (gfx1250) — compile-verified, matrix operand order checked by reading, not run on hardware
//
#include <hip/hip_runtime.h>
#include <hip/hip_bf16.h>
#include <math.h>

// ---------------------------------------------------------------------------
// MambaRegressor fused pipeline for MI455X (gfx1250, wave32, WMMA bf16).
//
// Shapes: B=32, K=512, Din=32, C=64, d_model=256, n_layer=4, patch_len=16,
// stride=8, di=512, ds=16, dconv=4, dtr=16, N=63, rows BC=2048.
//
// u (2048x63x256 f32, 132MB) lives in workspace -> L2-resident (192MB L2).
// Each layer is fully fused per row in one 512-thread block with 160KB LDS;
// all GEMMs use v_wmma_f32_16x16x32_bf16 with f32 accumulation.
// Weights are pre-swizzled into WMMA B-fragment-major layout so each B
// operand is one contiguous 32B per-lane read (2x global_load_b128).
// ---------------------------------------------------------------------------

typedef __bf16 bf16_t;
typedef __attribute__((ext_vector_type(16))) __bf16 v16bf;
typedef __attribute__((ext_vector_type(8)))  float  v8f;

#define NBATCH   32
#define KLEN     512
#define DIN      32
#define NCH      64          // C
#define DM       256         // d_model
#define NLAYER   4
#define PLEN     16
#define STRIDE   8
#define DI       512
#define DSTATE   16
#define DCONV    4
#define DTR      16
#define NT       63          // sequence length N
#define NTP      64          // padded
#define BC       2048        // NBATCH * NCH

static __device__ __forceinline__ bf16_t f2bf(float f) { return (bf16_t)f; }
static __device__ __forceinline__ float  bf2f(bf16_t h) { return (float)h; }
static __device__ __forceinline__ float  siluf(float v) { return v / (1.f + expf(-v)); }

// ---- WMMA fragment helpers (16x16x32 bf16, f32 accum) ----------------------
// A (16x32, MxK) per-lane layout: lane<16 -> M=lane, K in {k0..+7, k0+16..+23}
//                                 lane>=16 -> M=lane-16, K in {k0+8.., k0+24..}
static __device__ __forceinline__ v16bf load_a_lds(const bf16_t* base, int ld,
                                                   int m0, int k0, int lane) {
  int l = lane & 15, hi = (lane >> 4) & 1;
  const bf16_t* row = base + (size_t)(m0 + l) * ld + k0;
  v16bf a;
#pragma unroll
  for (int i = 0; i < 8; ++i) a[i] = row[hi * 8 + i];
#pragma unroll
  for (int i = 0; i < 8; ++i) a[8 + i] = row[16 + hi * 8 + i];
  return a;
}

// B fragments are pre-swizzled: frag (kt, nt) stored as 32 lanes x 16 bf16,
// contiguous.  One aligned 32B load per lane.
static __device__ __forceinline__ v16bf load_b_swz(const bf16_t* __restrict__ B,
                                                   int numNT, int kt, int nt,
                                                   int lane) {
  const bf16_t* p = B + ((((size_t)kt * numNT + nt) * 32 + lane) << 4);
  return *(const v16bf*)p;
}

static __device__ __forceinline__ v8f wmma_bf16(v16bf a, v16bf b, v8f c) {
  return __builtin_amdgcn_wmma_f32_16x16x32_bf16(false, a, false, b,
                                                 (short)0, c, false, false);
}

// ---------------------------------------------------------------------------
// f32 (K x N, row-major) -> bf16 swizzled B-fragment layout.
// dst[((kt*numNT + nt)*32 + lane)*16 + i] = src[k][n],
//   k = kt*32 + (i<8 ? hi*8+i : 16 + hi*8 + (i-8)),  hi = lane>>4
//   n = nt*16 + (lane&15)
// Rows k >= Ksrc are zero (K padding, e.g. pe_W 16 -> 32).
// ---------------------------------------------------------------------------
__global__ void cvt_swz_kernel(const float* __restrict__ src,
                               bf16_t* __restrict__ dst,
                               int Ksrc, int Kpad, int N) {
  int idx = blockIdx.x * blockDim.x + threadIdx.x;
  if (idx >= Kpad * N) return;
  int i    = idx & 15;
  int lane = (idx >> 4) & 31;
  int tile = idx >> 9;
  int numNT = N >> 4;
  int nt = tile % numNT;
  int kt = tile / numNT;
  int hi = lane >> 4;
  int krow = (i < 8) ? (hi * 8 + i) : (16 + hi * 8 + (i - 8));
  int k = kt * 32 + krow;
  int n = nt * 16 + (lane & 15);
  float v = (k < Ksrc) ? src[(size_t)k * N + n] : 0.f;
  dst[idx] = f2bf(v);
}

// ---------------------------------------------------------------------------
// Front-end: proj + LayerNorm + patch extract + patch embed -> u
// grid = (63, 32), block = 256 (8 waves)
// ---------------------------------------------------------------------------
__global__ void __launch_bounds__(256) frontend_kernel(
    const float* __restrict__ x,      // (32,512,32)
    const float* __restrict__ projW,  // (32,64)
    const float* __restrict__ projB,  // (64)
    const float* __restrict__ lng,    // (64)
    const float* __restrict__ lnb,    // (64)
    const bf16_t* __restrict__ peW,   // swizzled (Kpad=32, N=256)
    const float* __restrict__ peB,    // (256)
    float* __restrict__ u)            // (2048,63,256)
{
  __shared__ float  s_xs[PLEN * DIN];
  __shared__ float  s_h[PLEN * NCH];
  __shared__ float  s_mu[PLEN], s_inv[PLEN];
  __shared__ bf16_t s_A[NCH * 32];        // A matrix 64x32 (K zero-padded)

  const int tid = threadIdx.x, lane = tid & 31, wave = tid >> 5;
  const int n = blockIdx.x;   // patch 0..62
  const int b = blockIdx.y;   // batch 0..31

  for (int i = tid; i < PLEN * DIN; i += 256) {
    int p = i >> 5, j = i & 31;
    s_xs[i] = x[((size_t)b * KLEN + n * STRIDE + p) * DIN + j];
  }
  __syncthreads();

  for (int i = tid; i < PLEN * NCH; i += 256) {
    int p = i >> 6, c = i & 63;
    float acc = projB[c];
#pragma unroll
    for (int j = 0; j < DIN; ++j) acc += s_xs[p * DIN + j] * projW[j * NCH + c];
    s_h[i] = acc;
  }
  __syncthreads();

  if (tid < PLEN) {
    float mu = 0.f;
    for (int c = 0; c < NCH; ++c) mu += s_h[tid * NCH + c];
    mu *= (1.f / NCH);
    float var = 0.f;
    for (int c = 0; c < NCH; ++c) { float d = s_h[tid * NCH + c] - mu; var += d * d; }
    var *= (1.f / NCH);
    s_mu[tid] = mu;
    s_inv[tid] = rsqrtf(var + 1e-5f);
  }
  __syncthreads();

  for (int i = tid; i < NCH * 32; i += 256) {
    int c = i >> 5, p = i & 31;
    float v = 0.f;
    if (p < PLEN) v = (s_h[p * NCH + c] - s_mu[p]) * s_inv[p] * lng[c] + lnb[c];
    s_A[c * 32 + p] = f2bf(v);
  }
  __syncthreads();

  // u = A(64x16) @ peW(16x256) + peB : 4 m-tiles x 16 n-tiles
  for (int tile = wave; tile < 64; tile += 8) {
    int m0 = (tile >> 4) * 16, nt = tile & 15;
    v16bf a = load_a_lds(s_A, 32, m0, 0, lane);
    v16bf bb = load_b_swz(peW, 16, 0, nt, lane);
    v8f acc = {};
    acc = wmma_bf16(a, bb, acc);
    int l = lane & 15, hi = (lane >> 4) & 1;
    int col = nt * 16 + l;
#pragma unroll
    for (int j = 0; j < 8; ++j) {
      int c = m0 + j + hi * 8;
      u[(((size_t)b * NCH + c) * NT + n) * DM + col] = acc[j] + peB[col];
    }
  }
}

// ---------------------------------------------------------------------------
// Fused Mamba block, one sequence row per block. 512 threads (16 waves),
// 160KB static LDS.
// ---------------------------------------------------------------------------
__global__ void __launch_bounds__(512) layer_kernel(
    float* __restrict__ u,                 // (2048,63,256), residual stream
    const bf16_t* __restrict__ inW,        // swizzled (256,1024)
    const float*  __restrict__ convW,      // (512,4)
    const float*  __restrict__ convB,      // (512)
    const bf16_t* __restrict__ xprojW,     // swizzled (512,48)
    const float*  __restrict__ dtW,        // (16,512)
    const float*  __restrict__ dtB,        // (512)
    const float*  __restrict__ Alog,       // (512,16)
    const float*  __restrict__ Dskip,      // (512)
    const bf16_t* __restrict__ outW,       // swizzled (512,256)
    const float*  __restrict__ rmsw)       // (256)
{
  // region0: rms(u) A-matrix (64x256 bf16 = 32KB); reused as dbc (64x48 f32)
  __shared__ __align__(16) char s_r0[NTP * DM * 2];
  // region1: raw x (in-proj, 64x512 bf16), conv'd in place -> xc
  __shared__ __align__(16) bf16_t s_x[NTP * DI];
  // region2: silu(z) (64x512 bf16), overwritten in place by y*silu(z)
  __shared__ __align__(16) bf16_t s_z[NTP * DI];

  bf16_t* s_a   = (bf16_t*)s_r0;
  float*  s_dbc = (float*)s_r0;

  const int tid = threadIdx.x;
  const int lane = tid & 31;
  const int wave = tid >> 5;               // 0..15
  float* urow = u + (size_t)blockIdx.x * NT * DM;

  // ---- Phase 1: RMSNorm(u_row) -> bf16 A matrix (row 63 zero-padded) ------
  // Each lane owns 8 contiguous elements -> b128 global loads + b128 LDS store
  for (int t = wave; t < NTP; t += 16) {
    if (t < NT) {
      const float* ur = urow + (size_t)t * DM + lane * 8;
      float vals[8];
      float ss = 0.f;
#pragma unroll
      for (int i = 0; i < 8; ++i) { float v = ur[i]; vals[i] = v; ss += v * v; }
#pragma unroll
      for (int o = 16; o > 0; o >>= 1) ss += __shfl_xor(ss, o, 32);
      float sc = rsqrtf(ss * (1.f / DM) + 1e-5f);
#pragma unroll
      for (int i = 0; i < 8; ++i)
        s_a[t * DM + lane * 8 + i] = f2bf(vals[i] * rmsw[lane * 8 + i] * sc);
    } else {
#pragma unroll
      for (int i = 0; i < 8; ++i) s_a[t * DM + lane * 8 + i] = f2bf(0.f);
    }
  }
  __syncthreads();

  // ---- Phase 2: xz = rms(u)(64x256) @ inW(256x1024), WMMA ------------------
  for (int tile = wave; tile < 4 * 64; tile += 16) {
    int m0 = (tile >> 6) * 16, nt = tile & 63;
    v8f acc = {};
#pragma unroll
    for (int kk = 0; kk < 8; ++kk) {
      v16bf a = load_a_lds(s_a, DM, m0, kk * 32, lane);
      v16bf b = load_b_swz(inW, 64, kk, nt, lane);
      acc = wmma_bf16(a, b, acc);
    }
    int l = lane & 15, hi = (lane >> 4) & 1;
    int col = nt * 16 + l;
#pragma unroll
    for (int j = 0; j < 8; ++j) {
      int m = m0 + j + hi * 8;
      float v = acc[j];
      if (col < DI) s_x[m * DI + col] = f2bf(v);                // raw x
      else          s_z[m * DI + (col - DI)] = f2bf(siluf(v));  // silu(z)
    }
  }
  __syncthreads();

  // ---- Phase 3: depthwise causal conv (dconv=4) + bias + silu, in place ----
  {
    const int d = tid;  // one thread per channel
    float w0 = convW[d * DCONV + 0], w1 = convW[d * DCONV + 1];
    float w2 = convW[d * DCONV + 2], w3 = convW[d * DCONV + 3];
    float bb = convB[d];
    float xm3 = 0.f, xm2 = 0.f, xm1 = 0.f;
    for (int t = 0; t < NT; ++t) {
      float x0 = bf2f(s_x[t * DI + d]);
      float v = w0 * xm3 + w1 * xm2 + w2 * xm1 + w3 * x0 + bb;
      s_x[t * DI + d] = f2bf(siluf(v));
      xm3 = xm2; xm2 = xm1; xm1 = x0;
    }
    s_x[NT * DI + d] = f2bf(0.f);   // pad row
    s_z[NT * DI + d] = f2bf(0.f);   // pad row (Y buffer)
  }
  __syncthreads();

  // ---- Phase 4: dbc = xc(64x512) @ xprojW(512x48) -> f32 (overlaps s_a) ----
  for (int tile = wave; tile < 4 * 3; tile += 16) {
    int m0 = (tile / 3) * 16, nt = tile % 3;
    v8f acc = {};
#pragma unroll
    for (int kk = 0; kk < 16; ++kk) {
      v16bf a = load_a_lds(s_x, DI, m0, kk * 32, lane);
      v16bf b = load_b_swz(xprojW, 3, kk, nt, lane);
      acc = wmma_bf16(a, b, acc);
    }
    int l = lane & 15, hi = (lane >> 4) & 1;
#pragma unroll
    for (int j = 0; j < 8; ++j)
      s_dbc[(m0 + j + hi * 8) * 48 + nt * 16 + l] = acc[j];
  }
  __syncthreads();

  // ---- Phase 5: selective scan, one thread per channel d -------------------
  {
    const int d = tid;
    float A[DSTATE], h[DSTATE], dtw[DTR];
#pragma unroll
    for (int s = 0; s < DSTATE; ++s) { A[s] = -expf(Alog[d * DSTATE + s]); h[s] = 0.f; }
#pragma unroll
    for (int r = 0; r < DTR; ++r) dtw[r] = dtW[r * DI + d];
    const float dtb = dtB[d], dsk = Dskip[d];

    for (int t = 0; t < NT; ++t) {
      const float* db = s_dbc + t * 48;   // [0:16) dt_in, [16:32) B, [32:48) C
      float dtv = dtb;
#pragma unroll
      for (int r = 0; r < DTR; ++r) dtv += db[r] * dtw[r];
      dtv = (dtv > 20.f) ? dtv : log1pf(expf(dtv));   // softplus
      float xt = bf2f(s_x[t * DI + d]);
      float dx = dtv * xt;
      float y = 0.f;
#pragma unroll
      for (int s = 0; s < DSTATE; ++s) {
        h[s] = expf(dtv * A[s]) * h[s] + dx * db[16 + s];
        y += h[s] * db[32 + s];
      }
      y += xt * dsk;
      float zs = bf2f(s_z[t * DI + d]);     // silu(z)
      s_z[t * DI + d] = f2bf(y * zs);       // overwrite with gated output
    }
  }
  __syncthreads();

  // ---- Phase 6: du = Y(64x512) @ outW(512x256), residual add into u --------
  for (int tile = wave; tile < 4 * 16; tile += 16) {
    int m0 = (tile >> 4) * 16, nt = tile & 15;
    v8f acc = {};
#pragma unroll
    for (int kk = 0; kk < 16; ++kk) {
      v16bf a = load_a_lds(s_z, DI, m0, kk * 32, lane);
      v16bf b = load_b_swz(outW, 16, kk, nt, lane);
      acc = wmma_bf16(a, b, acc);
    }
    int l = lane & 15, hi = (lane >> 4) & 1;
    int col = nt * 16 + l;
#pragma unroll
    for (int j = 0; j < 8; ++j) {
      int m = m0 + j + hi * 8;
      if (m < NT) urow[(size_t)m * DM + col] += acc[j];
    }
  }
}

// ---------------------------------------------------------------------------
// Head: final RMS fused with the (63*256 x 1) head matvec. One block per row.
// ---------------------------------------------------------------------------
__global__ void __launch_bounds__(256) head_kernel(
    const float* __restrict__ u, const float* __restrict__ frw,
    const float* __restrict__ headW, const float* __restrict__ headB,
    float* __restrict__ feats)
{
  __shared__ float s_part[8];
  __shared__ float s_scale;
  const int tid = threadIdx.x, lane = tid & 31, wave = tid >> 5;
  const float* ur = u + (size_t)blockIdx.x * NT * DM;
  const float wgt = frw[tid];
  float acc = 0.f;

  for (int t = 0; t < NT; ++t) {
    float v = ur[t * DM + tid];
    float ss = v * v;
#pragma unroll
    for (int o = 16; o > 0; o >>= 1) ss += __shfl_xor(ss, o, 32);
    if (lane == 0) s_part[wave] = ss;
    __syncthreads();
    if (tid == 0) {
      float tot = 0.f;
      for (int w = 0; w < 8; ++w) tot += s_part[w];
      s_scale = rsqrtf(tot * (1.f / DM) + 1e-5f);
    }
    __syncthreads();
    acc += v * wgt * s_scale * headW[t * DM + tid];
    __syncthreads();
  }
#pragma unroll
  for (int o = 16; o > 0; o >>= 1) acc += __shfl_xor(acc, o, 32);
  if (lane == 0) s_part[wave] = acc;
  __syncthreads();
  if (tid == 0) {
    float tot = 0.f;
    for (int w = 0; w < 8; ++w) tot += s_part[w];
    feats[blockIdx.x] = tot + headB[0];
  }
}

// ---------------------------------------------------------------------------
// Final tiny heads: pos = feats@posW+posB, next = feats@nextW+nextB
// d_out = [pos (32x2) | next (32x2)], 128 floats. 1 block, 128 threads.
// ---------------------------------------------------------------------------
__global__ void final_kernel(const float* __restrict__ feats,
                             const float* __restrict__ posW, const float* __restrict__ posB,
                             const float* __restrict__ nextW, const float* __restrict__ nextB,
                             float* __restrict__ out)
{
  const int tid = threadIdx.x;         // 0..127
  const int half = tid >> 6;           // 0=pos, 1=next
  const int i = tid & 63;              // b*2 + j
  const int b = i >> 1, j = i & 1;
  const float* W = half ? nextW : posW;
  const float* Bv = half ? nextB : posB;
  float acc = Bv[j];
  for (int c = 0; c < NCH; ++c) acc += feats[b * NCH + c] * W[c * 2 + j];
  out[half * 64 + i] = acc;
}

// ---------------------------------------------------------------------------
extern "C" void kernel_launch(void* const* d_in, const int* in_sizes, int n_in,
                              void* d_out, int out_size, void* d_ws, size_t ws_size,
                              hipStream_t stream) {
  (void)in_sizes; (void)n_in; (void)out_size; (void)ws_size;

  const float* x       = (const float*)d_in[0];
  const float* projW   = (const float*)d_in[1];
  const float* projB   = (const float*)d_in[2];
  const float* lng     = (const float*)d_in[3];
  const float* lnb     = (const float*)d_in[4];
  const float* peW     = (const float*)d_in[5];
  const float* peB     = (const float*)d_in[6];
  const float* inW     = (const float*)d_in[7];
  const float* convW   = (const float*)d_in[8];
  const float* convB   = (const float*)d_in[9];
  const float* xprojW  = (const float*)d_in[10];
  const float* dtW     = (const float*)d_in[11];
  const float* dtB     = (const float*)d_in[12];
  const float* Alog    = (const float*)d_in[13];
  const float* Dskip   = (const float*)d_in[14];
  const float* outW    = (const float*)d_in[15];
  const float* rmsw    = (const float*)d_in[16];
  const float* frw     = (const float*)d_in[17];
  const float* headW   = (const float*)d_in[18];
  const float* headB   = (const float*)d_in[19];
  const float* posW    = (const float*)d_in[20];
  const float* posB    = (const float*)d_in[21];
  const float* nextW   = (const float*)d_in[22];
  const float* nextB   = (const float*)d_in[23];
  float* out = (float*)d_out;

  // workspace layout (all offsets 32B-aligned for v16bf loads)
  char* w = (char*)d_ws;
  size_t off = 0;
  float* u = (float*)(w + off);           off += (size_t)BC * NT * DM * 4;       // 132.1 MB
  float* feats = (float*)(w + off);       off += (size_t)BC * 4 * 2;
  bf16_t* inW_bf = (bf16_t*)(w + off);    off += (size_t)NLAYER * DM * 2 * DI * 2;
  bf16_t* outW_bf = (bf16_t*)(w + off);   off += (size_t)NLAYER * DI * DM * 2;
  bf16_t* xprojW_bf = (bf16_t*)(w + off); off += (size_t)NLAYER * DI * 48 * 2;
  bf16_t* peW_bf = (bf16_t*)(w + off);    off += (size_t)32 * DM * 2;

  // weight conversions f32 -> bf16, swizzled to WMMA B-fragment layout
  for (int l = 0; l < NLAYER; ++l) {
    int n;
    n = DM * (2 * DI);   // 256 x 1024
    cvt_swz_kernel<<<(n + 255) / 256, 256, 0, stream>>>(
        inW + (size_t)l * n, inW_bf + (size_t)l * n, DM, DM, 2 * DI);
    n = DI * DM;         // 512 x 256
    cvt_swz_kernel<<<(n + 255) / 256, 256, 0, stream>>>(
        outW + (size_t)l * n, outW_bf + (size_t)l * n, DI, DI, DM);
    n = DI * 48;         // 512 x 48
    cvt_swz_kernel<<<(n + 255) / 256, 256, 0, stream>>>(
        xprojW + (size_t)l * n, xprojW_bf + (size_t)l * n, DI, DI, 48);
  }
  {
    int n = 32 * DM;     // pe_W: 16 x 256, K zero-padded to 32
    cvt_swz_kernel<<<(n + 255) / 256, 256, 0, stream>>>(peW, peW_bf, PLEN, 32, DM);
  }

  // front-end: u = patch_embed(LN(x @ projW))
  {
    dim3 grid(NT, NBATCH);
    frontend_kernel<<<grid, 256, 0, stream>>>(x, projW, projB, lng, lnb,
                                              peW_bf, peB, u);
  }

  // 4 fused Mamba layers, one row per block
  for (int l = 0; l < NLAYER; ++l) {
    layer_kernel<<<BC, 512, 0, stream>>>(
        u,
        inW_bf    + (size_t)l * DM * 2 * DI,
        convW     + (size_t)l * DI * DCONV,
        convB     + (size_t)l * DI,
        xprojW_bf + (size_t)l * DI * 48,
        dtW       + (size_t)l * DTR * DI,
        dtB       + (size_t)l * DI,
        Alog      + (size_t)l * DI * DSTATE,
        Dskip     + (size_t)l * DI,
        outW_bf   + (size_t)l * DI * DM,
        rmsw      + (size_t)l * DM);
  }

  // head: final RMS + (63*256 x 1) matvec per row
  head_kernel<<<BC, 256, 0, stream>>>(u, frw, headW, headB, feats);

  // tiny output heads
  final_kernel<<<1, 128, 0, stream>>>(feats, posW, posB, nextW, nextB, out);
}
